// MsaPairWeightedAveraging_80178449482187
// MI455X (gfx1250) — matrix operand order repaired and checked by reading, not compiled, not run on hardware
//
#include <hip/hip_runtime.h>

// ---------------- CDNA5 WMMA types & helpers ----------------
typedef __attribute__((ext_vector_type(16))) __bf16 v16bf;
typedef __attribute__((ext_vector_type(8)))  float  v8f;

union FragBF { v16bf v; unsigned int u[8]; };

static __device__ inline unsigned short f32_to_bf16(float f) {
  unsigned int u = __float_as_uint(f);
  u += 0x7fffu + ((u >> 16) & 1u);   // round-to-nearest-even
  return (unsigned short)(u >> 16);
}
static __device__ inline float bf16_to_f32(unsigned short h) {
  return __uint_as_float(((unsigned int)h) << 16);
}

// CDNA5 async DMA: global -> LDS, 16 bytes per lane, tracked by ASYNCcnt.
static __device__ inline void async_b128(void* lds_ptr, const void* gptr) {
  unsigned loff = (unsigned)(size_t)lds_ptr;  // low 32 bits of LDS-aperture addr = LDS byte offset
  asm volatile("global_load_async_to_lds_b128 %0, %1, off"
               :: "v"(loff), "v"(gptr) : "memory");
}
static __device__ inline void wait_async0() {
  asm volatile("s_wait_asynccnt 0x0" ::: "memory");
}

// A fragment: 16x32 (MxK) bf16 from LDS laid out [m][k] (k contiguous, stride even)
static __device__ inline v16bf load_a_frag(const unsigned short* lds, int row_stride,
                                           int m_base, int k_base, int lane) {
  FragBF f;
  const int grp = (lane >> 4) & 1;
  const unsigned short* rowp = lds + (m_base + (lane & 15)) * row_stride;
#pragma unroll
  for (int v = 0; v < 8; ++v) {
    int k = k_base + (grp << 4) + (v << 1);
    f.u[v] = *(const unsigned int*)(rowp + k);
  }
  return f.v;
}

// B fragment: 32x16 (KxN) bf16 from LDS laid out [n][k] (k contiguous, stride even)
static __device__ inline v16bf load_b_frag(const unsigned short* lds, int row_stride,
                                           int n_base, int k_base, int lane) {
  FragBF f;
  const unsigned short* rowp = lds + (n_base + (lane & 15)) * row_stride;
  const int koff = k_base + ((lane >> 4) << 4);
#pragma unroll
  for (int v = 0; v < 8; ++v)
    f.u[v] = *(const unsigned int*)(rowp + koff + (v << 1));
  return f.v;
}

static __device__ inline v8f wmma_bf16(v16bf a, v16bf b, v8f c) {
  return __builtin_amdgcn_wmma_f32_16x16x32_bf16(false, a, false, b, (short)0, c,
                                                 false, false);
}

#define S_DIM 512
#define N_DIM 384

// ============================================================
// Kernel 0: one-time weight prep.
// W_vg [64][512] f32 -> wvg_t [n=512][k=64] bf16 (K-contiguous B layout)
// W_out[256][64] f32 -> wout_t [c=64][k=256] bf16
// ============================================================
__global__ __launch_bounds__(256) void k_prep(
    const float* __restrict__ Wvg, const float* __restrict__ Wout,
    unsigned short* __restrict__ wvg_t, unsigned short* __restrict__ wout_t) {
  int idx = blockIdx.x * 256 + threadIdx.x;
  if (idx < 64 * 512) {
    int k = idx >> 9, n = idx & 511;
    wvg_t[n * 64 + k] = f32_to_bf16(Wvg[idx]);
  } else {
    int j = idx - 64 * 512;  // < 256*64
    int k = j >> 6, c = j & 63;
    wout_t[c * 256 + k] = f32_to_bf16(Wout[j]);
  }
}

// ============================================================
// Kernel 1: LN(msa) -> bf16 -> GEMM vs W_vg -> vals/gates
// W_vg tile DMA'd async into LDS while LN runs.
// vals  layout: [h][s][d][n]   (K-contiguous B for the einsum)
// gates layout: [h][s][i][d]   (post-sigmoid)
// ============================================================
__global__ __launch_bounds__(256) void k_msa_vg(
    const float* __restrict__ msa, const float* __restrict__ lng,
    const float* __restrict__ lnb, const unsigned short* __restrict__ wvg_t,
    unsigned short* __restrict__ vals, unsigned short* __restrict__ gates) {
  __shared__ unsigned short w_lds[512 * 64];  // W_vg as [n][k] bf16
  __shared__ unsigned short a_lds[64 * 64];   // 64 LN'd rows, [m][k]
  __shared__ float red_s[64][4], red_q[64][4];
  __shared__ float mu_s[64], rs_s[64];

  const int tid = threadIdx.x, lane = tid & 31, wave = tid >> 5;

  // ---- async DMA W_vg bf16 [n][k] into LDS (overlaps with the LN below)
  {
    const uint4* ws4 = (const uint4*)wvg_t;
    uint4* wd4 = (uint4*)w_lds;
    for (int f = tid; f < 4096; f += 256) async_b128(wd4 + f, ws4 + f);
  }

  // ---- LayerNorm on 64 rows (4 threads per row, 16 f32 each)
  const int r = tid >> 2, q = tid & 3;
  const float* xrow = msa + ((size_t)blockIdx.x * 64 + r) * 64;
  float4 xv[4];
  float s1 = 0.f, s2 = 0.f;
#pragma unroll
  for (int j = 0; j < 4; ++j) {
    xv[j] = ((const float4*)xrow)[q * 4 + j];
    s1 += xv[j].x + xv[j].y + xv[j].z + xv[j].w;
    s2 += xv[j].x * xv[j].x + xv[j].y * xv[j].y + xv[j].z * xv[j].z + xv[j].w * xv[j].w;
  }
  red_s[r][q] = s1;
  red_q[r][q] = s2;
  __syncthreads();
  if (q == 0) {
    float t1 = red_s[r][0] + red_s[r][1] + red_s[r][2] + red_s[r][3];
    float t2 = red_q[r][0] + red_q[r][1] + red_q[r][2] + red_q[r][3];
    float mu = t1 * (1.0f / 64.0f);
    float var = t2 * (1.0f / 64.0f) - mu * mu;
    mu_s[r] = mu;
    rs_s[r] = rsqrtf(var + 1e-5f);
  }
  __syncthreads();
  {
    float mu = mu_s[r], rs = rs_s[r];
#pragma unroll
    for (int j = 0; j < 4; ++j) {
      int c = q * 16 + j * 4;
      float4 gg = ((const float4*)lng)[c >> 2];
      float4 bb = ((const float4*)lnb)[c >> 2];
      a_lds[r * 64 + c + 0] = f32_to_bf16((xv[j].x - mu) * rs * gg.x + bb.x);
      a_lds[r * 64 + c + 1] = f32_to_bf16((xv[j].y - mu) * rs * gg.y + bb.y);
      a_lds[r * 64 + c + 2] = f32_to_bf16((xv[j].z - mu) * rs * gg.z + bb.z);
      a_lds[r * 64 + c + 3] = f32_to_bf16((xv[j].w - mu) * rs * gg.w + bb.w);
    }
  }
  wait_async0();
  __syncthreads();

  // ---- GEMM: wave owns 64 cols; 4x4 tiles of 16x16, K=64 (2 wmma steps)
  const int n0 = wave * 64;
  v8f acc[4][4];
#pragma unroll
  for (int ti = 0; ti < 4; ++ti)
#pragma unroll
    for (int tc = 0; tc < 4; ++tc)
#pragma unroll
      for (int e = 0; e < 8; ++e) acc[ti][tc][e] = 0.f;

#pragma unroll
  for (int kk = 0; kk < 64; kk += 32) {
    v16bf af[4], bfr[4];
#pragma unroll
    for (int ti = 0; ti < 4; ++ti) af[ti] = load_a_frag(a_lds, 64, ti * 16, kk, lane);
#pragma unroll
    for (int tc = 0; tc < 4; ++tc) bfr[tc] = load_b_frag(w_lds, 64, n0 + tc * 16, kk, lane);
#pragma unroll
    for (int ti = 0; ti < 4; ++ti)
#pragma unroll
      for (int tc = 0; tc < 4; ++tc) acc[ti][tc] = wmma_bf16(af[ti], bfr[tc], acc[ti][tc]);
  }

  // ---- epilogue: split into values / sigmoid-gates
  const int rowoff = (lane >> 4) << 3, cl = lane & 15;
#pragma unroll
  for (int ti = 0; ti < 4; ++ti) {
#pragma unroll
    for (int tc = 0; tc < 4; ++tc) {
      int c = n0 + tc * 16 + cl;
#pragma unroll
      for (int rr = 0; rr < 8; ++rr) {
        int m = ti * 16 + rowoff + rr;
        int flat = blockIdx.x * 64 + m;
        int s = flat / N_DIM, n = flat - s * N_DIM;
        float v = acc[ti][tc][rr];
        if (c < 256) {
          int h = c >> 5, d = c & 31;
          vals[(((size_t)h * S_DIM + s) * 32 + d) * N_DIM + n] = f32_to_bf16(v);
        } else {
          int cp = c - 256, h = cp >> 5, d = cp & 31;
          float gv = 1.0f / (1.0f + __expf(-v));
          gates[(((size_t)h * S_DIM + s) * N_DIM + n) * 32 + d] = f32_to_bf16(gv);
        }
      }
    }
  }
}

// ============================================================
// Kernel 2: pair LN -> @W_b[128x8] -> mask -> softmax(j) -> wts bf16 [h][i][j]
// One block per i; one wave per j-row; wave w handles head w for softmax.
// ============================================================
__global__ __launch_bounds__(256) void k_pair_softmax(
    const float* __restrict__ pair, const unsigned char* __restrict__ mask,
    const float* __restrict__ lng, const float* __restrict__ lnb,
    const float* __restrict__ Wb, unsigned short* __restrict__ wts) {
  __shared__ float bias_s[8][384];
  __shared__ float wb_s[128 * 8];
  __shared__ float g_s[128], b_s[128];
  const int tid = threadIdx.x, lane = tid & 31, wave = tid >> 5;
  const int i = blockIdx.x;

  for (int idx = tid; idx < 1024; idx += 256) wb_s[idx] = Wb[idx];
  if (tid < 128) { g_s[tid] = lng[tid]; b_s[tid] = lnb[tid]; }
  __syncthreads();

  const float* pb = pair + (size_t)i * N_DIM * 128;
  for (int it = 0; it < 48; ++it) {
    int j = it * 8 + wave;
    float4 x = ((const float4*)(pb + (size_t)j * 128))[lane];
    float s1 = x.x + x.y + x.z + x.w;
    float s2 = x.x * x.x + x.y * x.y + x.z * x.z + x.w * x.w;
    for (int o = 16; o > 0; o >>= 1) { s1 += __shfl_xor(s1, o); s2 += __shfl_xor(s2, o); }
    float mu = s1 * (1.0f / 128.0f);
    float var = s2 * (1.0f / 128.0f) - mu * mu;
    float rs = rsqrtf(var + 1e-5f);
    int c0 = lane * 4;
    float xn0 = (x.x - mu) * rs * g_s[c0 + 0] + b_s[c0 + 0];
    float xn1 = (x.y - mu) * rs * g_s[c0 + 1] + b_s[c0 + 1];
    float xn2 = (x.z - mu) * rs * g_s[c0 + 2] + b_s[c0 + 2];
    float xn3 = (x.w - mu) * rs * g_s[c0 + 3] + b_s[c0 + 3];
    float acc[8];
#pragma unroll
    for (int h = 0; h < 8; ++h)
      acc[h] = xn0 * wb_s[(c0 + 0) * 8 + h] + xn1 * wb_s[(c0 + 1) * 8 + h] +
               xn2 * wb_s[(c0 + 2) * 8 + h] + xn3 * wb_s[(c0 + 3) * 8 + h];
#pragma unroll
    for (int h = 0; h < 8; ++h)
      for (int o = 16; o > 0; o >>= 1) acc[h] += __shfl_xor(acc[h], o);
    if (lane == 0) {
      bool mk = mask[j] != 0;
#pragma unroll
      for (int h = 0; h < 8; ++h) bias_s[h][j] = mk ? acc[h] : -3.402823466e38f;
    }
  }
  __syncthreads();

  const int h = wave;
  float m = -3.402823466e38f;
#pragma unroll
  for (int k = 0; k < 12; ++k) m = fmaxf(m, bias_s[h][lane + 32 * k]);
  for (int o = 16; o > 0; o >>= 1) m = fmaxf(m, __shfl_xor(m, o));
  float e[12], ssum = 0.f;
#pragma unroll
  for (int k = 0; k < 12; ++k) { e[k] = __expf(bias_s[h][lane + 32 * k] - m); ssum += e[k]; }
  for (int o = 16; o > 0; o >>= 1) ssum += __shfl_xor(ssum, o);
  float inv = 1.0f / ssum;
#pragma unroll
  for (int k = 0; k < 12; ++k)
    wts[((size_t)h * N_DIM + i) * N_DIM + lane + 32 * k] = f32_to_bf16(e[k] * inv);
}

// ============================================================
// Kernel 3: fused einsum (weights @ values) + gating + W_out projection.
// Double-buffered: head h+1 tiles are async-DMA'd into LDS while head h's
// 12 WMMA k-steps run. 8 waves; then fused 16-WMMA projection.
// ============================================================
__global__ __launch_bounds__(256) void k_attn_out(
    const unsigned short* __restrict__ wts, const unsigned short* __restrict__ vals,
    const unsigned short* __restrict__ gates, const unsigned short* __restrict__ wout_t,
    float* __restrict__ out) {
  __shared__ unsigned short a_lds[2 * 64 * 384];  // wts tile   [i][n], double-buffered
  __shared__ unsigned short b_lds[2 * 32 * 384];  // values     [d][n], double-buffered
  __shared__ unsigned short g_lds[64 * 256];      // gated out  [i][h*32+d]
  __shared__ unsigned short wout_lds[64 * 256];   // W_out as   [c][k]
  const int tid = threadIdx.x, lane = tid & 31, wave = tid >> 5;
  const int s = blockIdx.x / 6;
  const int i0 = (blockIdx.x % 6) * 64;
  const int rowoff = (lane >> 4) << 3, cl = lane & 15;
  const int tile_i = wave >> 1, tile_d = wave & 1;

  // async DMA: W_out bf16 [c][k] (prepped) straight into LDS
  {
    const uint4* ws4 = (const uint4*)wout_t;
    uint4* wd4 = (uint4*)wout_lds;
    for (int f = tid; f < 2048; f += 256) async_b128(wd4 + f, ws4 + f);
  }

  // async stage of one head's weight tile + value block
  auto issue_stage = [&](int h, int buf) {
    const uint4* as = (const uint4*)(wts + ((size_t)h * N_DIM + i0) * N_DIM);
    uint4* ad = (uint4*)(a_lds + buf * 64 * 384);
    for (int f = tid; f < 3072; f += 256) async_b128(ad + f, as + f);
    const uint4* bs = (const uint4*)(vals + ((size_t)h * S_DIM + s) * 32 * N_DIM);
    uint4* bd = (uint4*)(b_lds + buf * 32 * 384);
    for (int f = tid; f < 1536; f += 256) async_b128(bd + f, bs + f);
  };

  issue_stage(0, 0);
  wait_async0();
  __syncthreads();

  for (int h = 0; h < 8; ++h) {
    const int buf = h & 1;
    if (h < 7) issue_stage(h + 1, buf ^ 1);  // overlaps with WMMAs below

    const unsigned short* ab = a_lds + buf * 64 * 384;
    const unsigned short* bb = b_lds + buf * 32 * 384;
    v8f acc;
#pragma unroll
    for (int e = 0; e < 8; ++e) acc[e] = 0.f;
#pragma unroll
    for (int kk = 0; kk < 384; kk += 32) {
      v16bf af = load_a_frag(ab, 384, tile_i * 16, kk, lane);
      v16bf bfr = load_b_frag(bb, 384, tile_d * 16, kk, lane);
      acc = wmma_bf16(af, bfr, acc);
    }

    // gate + pack into G
    const unsigned short* gb = gates + ((size_t)h * S_DIM + s) * N_DIM * 32;
    const int d = tile_d * 16 + cl;
#pragma unroll
    for (int rr = 0; rr < 8; ++rr) {
      int im = tile_i * 16 + rowoff + rr;
      float gv = bf16_to_f32(gb[(size_t)(i0 + im) * 32 + d]);
      g_lds[im * 256 + h * 32 + d] = f32_to_bf16(acc[rr] * gv);
    }
    wait_async0();
    __syncthreads();
  }

  // projection: G[64x256] @ Wout[256x64] -> out f32
#pragma unroll
  for (int tt = 0; tt < 2; ++tt) {
    int t = wave + tt * 8;
    int ti = t >> 2, tc = t & 3;
    v8f acc;
#pragma unroll
    for (int e = 0; e < 8; ++e) acc[e] = 0.f;
#pragma unroll
    for (int kk = 0; kk < 256; kk += 32) {
      v16bf af = load_a_frag(g_lds, 256, ti * 16, kk, lane);
      v16bf bfr = load_b_frag(wout_lds, 256, tc * 16, kk, lane);
      acc = wmma_bf16(af, bfr, acc);
    }
#pragma unroll
    for (int rr = 0; rr < 8; ++rr) {
      int irow = i0 + ti * 16 + rowoff + rr;
      out[((size_t)s * N_DIM + irow) * 64 + tc * 16 + cl] = acc[rr];
    }
  }
}

// ============================================================
extern "C" void kernel_launch(void* const* d_in, const int* in_sizes, int n_in,
                              void* d_out, int out_size, void* d_ws, size_t ws_size,
                              hipStream_t stream) {
  const float* msa       = (const float*)d_in[0];
  const float* pair      = (const float*)d_in[1];
  const unsigned char* mask = (const unsigned char*)d_in[2];
  const float* ln_msa_g  = (const float*)d_in[3];
  const float* ln_msa_b  = (const float*)d_in[4];
  const float* Wvg       = (const float*)d_in[5];
  const float* ln_pair_g = (const float*)d_in[6];
  const float* ln_pair_b = (const float*)d_in[7];
  const float* Wb        = (const float*)d_in[8];
  const float* Wout      = (const float*)d_in[9];
  float* out = (float*)d_out;

  char* ws = (char*)d_ws;
  unsigned short* vals   = (unsigned short*)(ws);              // 8*512*384*32 bf16 = 100663296 B
  unsigned short* gates  = (unsigned short*)(ws + 100663296);  // same size
  unsigned short* wtsp   = (unsigned short*)(ws + 201326592);  // 8*384*384 bf16 = 2359296 B
  unsigned short* wvg_t  = (unsigned short*)(ws + 203685888);  // 512*64 bf16 = 65536 B
  unsigned short* wout_t = (unsigned short*)(ws + 203751424);  // 64*256 bf16 = 32768 B

  k_prep<<<192, 256, 0, stream>>>(Wvg, Wout, wvg_t, wout_t);
  k_msa_vg<<<3072, 256, 0, stream>>>(msa, ln_msa_g, ln_msa_b, wvg_t, vals, gates);
  k_pair_softmax<<<384, 256, 0, stream>>>(pair, mask, ln_pair_g, ln_pair_b, Wb, wtsp);
  k_attn_out<<<3072, 256, 0, stream>>>(wtsp, vals, gates, wout_t, out);
}
